// GCN_5669356832299
// MI455X (gfx1250) — matrix operand order repaired and checked by reading, not compiled, not run on hardware
//
#include <hip/hip_runtime.h>
#include <cstdint>

// ---------------------------------------------------------------------------
// GCN forward for MI455X (gfx1250): bf16 WMMA GEMM + L2-resident atomic
// scatter aggregation.  wave32, 16x16x32 bf16 WMMA with fp32 accumulate.
// ---------------------------------------------------------------------------

typedef __attribute__((ext_vector_type(16))) __bf16 v16bf;
typedef __attribute__((ext_vector_type(2)))  __bf16 vbf2;
typedef __attribute__((ext_vector_type(2)))  float  vf2;
typedef __attribute__((ext_vector_type(8)))  float  v8f;

// packed fp32x2 -> bf16x2 (lowers to v_cvt_pk_bf16_f32)
__device__ __forceinline__ unsigned int cvt2(float x, float y) {
  vf2 f = {x, y};
  return __builtin_bit_cast(unsigned int, __builtin_convertvector(f, vbf2));
}

__device__ __forceinline__ void cvt16(const float4 r[4], uint4 out[2]) {
  out[0] = make_uint4(cvt2(r[0].x, r[0].y), cvt2(r[0].z, r[0].w),
                      cvt2(r[1].x, r[1].y), cvt2(r[1].z, r[1].w));
  out[1] = make_uint4(cvt2(r[2].x, r[2].y), cvt2(r[2].z, r[2].w),
                      cvt2(r[3].x, r[3].y), cvt2(r[3].z, r[3].w));
}

// ---------------------------------------------------------------------------
// Z = H (MxK, fp32 row-major) * W^T (W is NxK fp32 row-major) -> Z (MxN fp32)
// 128x128 block tile, 8 waves, each wave: 16(M) x 128(N) via 8 WMMA tiles.
// K-step 32. Global->register prefetch of step k+1 overlaps the WMMAs of
// step k. All 8 B fragments are loaded before the WMMA burst so the in-order
// DS returns let the scheduler stagger s_wait_dscnt instead of draining to 0
// in front of every WMMA. LDS rows padded to 40 bf16 (80 B) for bank spread.
// ---------------------------------------------------------------------------
__global__ __launch_bounds__(256)
void k_gemm_bf16(const float* __restrict__ Hmat, const float* __restrict__ Wmat,
                 float* __restrict__ Zmat, int M, int N, int K) {
  __shared__ __attribute__((aligned(16))) unsigned short lA[128 * 40];
  __shared__ __attribute__((aligned(16))) unsigned short lB[128 * 40];

  const int m0   = blockIdx.x * 128;
  const int n0   = blockIdx.y * 128;
  const int t    = threadIdx.x;
  const int wave = t >> 5;
  const int lane = t & 31;
  const int r    = lane & 15;   // row (A) / col (B) within 16
  const int hi   = lane >> 4;   // K-half selector

  const int srow  = t >> 1;     // staging: row 0..127
  const int shalf = t & 1;      // staging: 16-float half of the 32-wide K slab

  // Per-thread staging sources (float4 granularity).
  const bool   va = (m0 + srow) < M;
  const float4* pa = reinterpret_cast<const float4*>(
      Hmat + (size_t)(va ? (m0 + srow) : 0) * K + shalf * 16);
  const float4* pb = reinterpret_cast<const float4*>(
      Wmat + (size_t)(n0 + srow) * K + shalf * 16);

  float4 ra[4], rb[4];
  auto prefetch = [&](int k0) {
    const int q0 = k0 >> 2;  // float4 index of k0 within the row
#pragma unroll
    for (int q = 0; q < 4; ++q) rb[q] = pb[q0 + q];
    if (va) {
#pragma unroll
      for (int q = 0; q < 4; ++q) ra[q] = pa[q0 + q];
    } else {
#pragma unroll
      for (int q = 0; q < 4; ++q) ra[q] = make_float4(0.f, 0.f, 0.f, 0.f);
    }
  };

  v8f acc[8];
#pragma unroll
  for (int j = 0; j < 8; ++j)
    acc[j] = (v8f){0.f, 0.f, 0.f, 0.f, 0.f, 0.f, 0.f, 0.f};

  uint4* const sA = reinterpret_cast<uint4*>(lA + srow * 40 + shalf * 16);
  uint4* const sB = reinterpret_cast<uint4*>(lB + srow * 40 + shalf * 16);
  const unsigned short* const arow  = lA + (16 * wave + r) * 40;
  const unsigned short* const bbase = lB + r * 40 + hi * 16;

  prefetch(0);
  for (int k0 = 0; k0 < K; k0 += 32) {
    uint4 sa[2], sb[2];
    cvt16(ra, sa);
    cvt16(rb, sb);
    __syncthreads();              // prior compute done reading LDS
    sA[0] = sa[0]; sA[1] = sa[1];
    sB[0] = sb[0]; sB[1] = sb[1];
    __syncthreads();              // tiles visible
    if (k0 + 32 < K) prefetch(k0 + 32);  // issue early; hides under WMMAs

    union Frag { uint4 q[2]; v16bf v; } fa, fb[8];
    // A fragment: row = 16*wave + r; K = hi*16 + 0..15 (contiguous 32B)
    fa.q[0] = *reinterpret_cast<const uint4*>(arow + hi * 16);
    fa.q[1] = *reinterpret_cast<const uint4*>(arow + hi * 16 + 8);
    // All 8 B fragments first (in-order DS returns -> staggered waits).
    // B fragment j: col = 16*j + r; K = hi*16 + 0..15 (contiguous 32B)
#pragma unroll
    for (int j = 0; j < 8; ++j) {
      const unsigned short* bn = bbase + j * 16 * 40;
      fb[j].q[0] = *reinterpret_cast<const uint4*>(bn);
      fb[j].q[1] = *reinterpret_cast<const uint4*>(bn + 8);
    }
#pragma unroll
    for (int j = 0; j < 8; ++j) {
      acc[j] = __builtin_amdgcn_wmma_f32_16x16x32_bf16(
          false, fa.v, false, fb[j].v, (short)0, acc[j], false, false);
    }
  }

  // Epilogue: D layout -> lanes 0-15: (M=v, N=lane); lanes 16-31: (M=v+8)
  const int orow0 = m0 + 16 * wave + hi * 8;
#pragma unroll
  for (int j = 0; j < 8; ++j) {
    const int col = n0 + 16 * j + r;
#pragma unroll
    for (int vv = 0; vv < 8; ++vv) {
      const int row = orow0 + vv;
      if (row < M) Zmat[(size_t)row * N + col] = acc[j][vv];
    }
  }
}

// ---------------------------------------------------------------------------
// Degree / normalization helpers
// ---------------------------------------------------------------------------
__global__ void k_fill(float* __restrict__ p, int n, float v) {
  int i = blockIdx.x * blockDim.x + threadIdx.x;
  if (i < n) p[i] = v;
}

__global__ void k_deg(const long long* __restrict__ dst, float* __restrict__ deg, int E) {
  int i = blockIdx.x * blockDim.x + threadIdx.x;
  if (i < E) atomicAdd(&deg[(int)dst[i]], 1.0f);
}

__global__ void k_rsqrt(float* __restrict__ p, int n) {
  int i = blockIdx.x * blockDim.x + threadIdx.x;
  if (i < n) p[i] = rsqrtf(p[i]);  // deg >= 1 (self-loop), no zero guard needed
}

__global__ void k_zero4(float4* __restrict__ p, size_t n4) {
  size_t i = (size_t)blockIdx.x * blockDim.x + threadIdx.x;
  if (i < n4) p[i] = make_float4(0.f, 0.f, 0.f, 0.f);
}

// ---------------------------------------------------------------------------
// Edge aggregation: one block per edge, F/4 threads, float4 gather + 4
// atomic adds (resolve in L2 -- whole Agg buffer is L2-resident).
// ---------------------------------------------------------------------------
__global__ void k_aggregate(const float4* __restrict__ z, float* __restrict__ agg,
                            const long long* __restrict__ src,
                            const long long* __restrict__ dst,
                            const float* __restrict__ dinv, int F4) {
  const int e = blockIdx.x;
  const int t = threadIdx.x;
  const long long s = src[e];
  const long long d = dst[e];
  const float w = dinv[s] * dinv[d];
  float4 v = z[(size_t)s * F4 + t];
  float* out = agg + ((size_t)d * F4 + t) * 4;
  atomicAdd(out + 0, v.x * w);
  atomicAdd(out + 1, v.y * w);
  atomicAdd(out + 2, v.z * w);
  atomicAdd(out + 3, v.w * w);
}

// ---------------------------------------------------------------------------
// Epilogue: out = agg + Z*dinv^2 (self-loop) + bias, optional ReLU.
// ---------------------------------------------------------------------------
__global__ void k_finish(const float4* __restrict__ agg, const float4* __restrict__ z,
                         const float* __restrict__ dinv, const float* __restrict__ bias,
                         float4* __restrict__ out, int M, int F4, int relu) {
  size_t i = (size_t)blockIdx.x * blockDim.x + threadIdx.x;
  if (i >= (size_t)M * F4) return;
  const int row = (int)(i / F4);
  const int c4 = (int)(i % F4) * 4;
  float s = dinv[row];
  s *= s;
  float4 a = agg[i], zz = z[i], r;
  r.x = a.x + zz.x * s + bias[c4 + 0];
  r.y = a.y + zz.y * s + bias[c4 + 1];
  r.z = a.z + zz.z * s + bias[c4 + 2];
  r.w = a.w + zz.w * s + bias[c4 + 3];
  if (relu) {
    r.x = fmaxf(r.x, 0.f); r.y = fmaxf(r.y, 0.f);
    r.z = fmaxf(r.z, 0.f); r.w = fmaxf(r.w, 0.f);
  }
  out[i] = r;
}

// ---------------------------------------------------------------------------
// Row softmax, one wave (32 lanes) per row, 256 cols -> 8/lane. wave32 shfl.
// ---------------------------------------------------------------------------
__global__ void k_softmax(const float* __restrict__ logits, float* __restrict__ out, int F) {
  const int row = blockIdx.x;
  const int lane = threadIdx.x;
  const float* p = logits + (size_t)row * F;
  float v[8];
  float mx = -3.4e38f;
#pragma unroll
  for (int i = 0; i < 8; ++i) {
    v[i] = p[lane + 32 * i];
    mx = fmaxf(mx, v[i]);
  }
#pragma unroll
  for (int off = 16; off > 0; off >>= 1) mx = fmaxf(mx, __shfl_xor(mx, off, 32));
  float sum = 0.f;
#pragma unroll
  for (int i = 0; i < 8; ++i) {
    v[i] = __expf(v[i] - mx);
    sum += v[i];
  }
#pragma unroll
  for (int off = 16; off > 0; off >>= 1) sum += __shfl_xor(sum, off, 32);
  const float inv = 1.0f / sum;
#pragma unroll
  for (int i = 0; i < 8; ++i) out[(size_t)row * F + lane + 32 * i] = v[i] * inv;
}

// ---------------------------------------------------------------------------
static void run_layer(const float* Hin, const float* W, const float* b,
                      float* Zbuf, float* Abuf, float* Hout,
                      int M, int N, int K,
                      const long long* esrc, const long long* edst,
                      const float* dinv, int E, int relu, hipStream_t stream) {
  dim3 g((M + 127) / 128, N / 128);
  k_gemm_bf16<<<g, 256, 0, stream>>>(Hin, W, Zbuf, M, N, K);
  const size_t tot4 = (size_t)M * (N / 4);
  k_zero4<<<(unsigned)((tot4 + 255) / 256), 256, 0, stream>>>((float4*)Abuf, tot4);
  k_aggregate<<<E, N / 4, 0, stream>>>((const float4*)Zbuf, Abuf, esrc, edst, dinv, N / 4);
  k_finish<<<(unsigned)((tot4 + 255) / 256), 256, 0, stream>>>(
      (const float4*)Abuf, (const float4*)Zbuf, dinv, b, (float4*)Hout, M, N / 4, relu);
}

extern "C" void kernel_launch(void* const* d_in, const int* in_sizes, int n_in,
                              void* d_out, int out_size, void* d_ws, size_t ws_size,
                              hipStream_t stream) {
  (void)n_in; (void)out_size; (void)ws_size;

  const float*     x    = (const float*)d_in[0];
  const long long* eidx = (const long long*)d_in[1];  // edge_idx is int64 in reference
  const float* W1 = (const float*)d_in[2]; const float* b1 = (const float*)d_in[3];
  const float* W2 = (const float*)d_in[4]; const float* b2 = (const float*)d_in[5];
  const float* W3 = (const float*)d_in[6]; const float* b3 = (const float*)d_in[7];
  const float* W4 = (const float*)d_in[8]; const float* b4 = (const float*)d_in[9];

  const int D_IN = 512, D_HID = 512, D_OUT = 256;
  const int M = in_sizes[0] / D_IN;   // 20000 nodes
  const int E = in_sizes[1] / 2;      // 320000 edges
  const long long* esrc = eidx;       // edge_idx[0]
  const long long* edst = eidx + E;   // edge_idx[1]

  // Workspace layout (fp32):  dinv | H | Z | Agg  (~123 MB total)
  float* ws   = (float*)d_ws;
  float* dinv = ws;                                  // M (padded to 20480)
  float* Hbuf = ws + 20480;                          // M * 512
  float* Zbuf = Hbuf + (size_t)M * D_HID;            // M * 512
  float* Abuf = Zbuf + (size_t)M * D_HID;            // M * 512

  // deg[i] = 1 (self-loop) + in-degree ; dinv = rsqrt(deg)
  k_fill<<<(M + 255) / 256, 256, 0, stream>>>(dinv, M, 1.0f);
  k_deg<<<(E + 255) / 256, 256, 0, stream>>>(edst, dinv, E);
  k_rsqrt<<<(M + 255) / 256, 256, 0, stream>>>(dinv, M);

  run_layer(x,    W1, b1, Zbuf, Abuf, Hbuf, M, D_HID, D_IN,  esrc, edst, dinv, E, 1, stream);
  run_layer(Hbuf, W2, b2, Zbuf, Abuf, Hbuf, M, D_HID, D_HID, esrc, edst, dinv, E, 1, stream);
  run_layer(Hbuf, W3, b3, Zbuf, Abuf, Hbuf, M, D_HID, D_HID, esrc, edst, dinv, E, 1, stream);
  // Layer 4: logits (no ReLU) written in-place into Abuf, then softmax -> d_out
  run_layer(Hbuf, W4, b4, Zbuf, Abuf, Abuf, M, D_OUT, D_HID, esrc, edst, dinv, E, 0, stream);
  k_softmax<<<M, 32, 0, stream>>>(Abuf, (float*)d_out, D_OUT);
}
